// MambaBlock2D_41850161332281
// MI455X (gfx1250) — matrix operand order repaired and checked by reading, not compiled, not run on hardware
//
#include <hip/hip_runtime.h>
#include <hip/hip_bf16.h>

typedef __attribute__((ext_vector_type(16))) _Float16 v16h;
typedef __attribute__((ext_vector_type(8)))  _Float16 v8h;
typedef __attribute__((ext_vector_type(8)))  float    v8f;
typedef __attribute__((ext_vector_type(4)))  int      v4i;

#define B_SZ     2
#define C_DIM    256
#define HW       4096
#define L_SEQ    4096
#define M_ROWS   (B_SZ * L_SEQ)   // 8192
#define D_INNER_ 512
#define D_STATE_ 16

#define AS1 __attribute__((address_space(1)))
#define AS3 __attribute__((address_space(3)))

#if __has_builtin(__builtin_amdgcn_global_load_async_to_lds_b128)
#define HAVE_ASYNC_LDS 1
#else
#define HAVE_ASYNC_LDS 0
#endif

__device__ __forceinline__ void wait_async0() {
#if __has_builtin(__builtin_amdgcn_s_wait_asynccnt)
    __builtin_amdgcn_s_wait_asynccnt(0);
#elif HAVE_ASYNC_LDS
    asm volatile("s_wait_asynccnt 0x0" ::: "memory");
#endif
}

// ---------------------------------------------------------------------------
// f32 -> f16 weight conversion
// ---------------------------------------------------------------------------
__global__ void k_f32_to_f16(const float* __restrict__ src, _Float16* __restrict__ dst, int n) {
    int i = blockIdx.x * 256 + threadIdx.x;
    if (i < n) dst[i] = (_Float16)src[i];
}

// dt_proj_w (512 x 16) -> f16 padded (512 x 32)
__global__ void k_pad_dtw(const float* __restrict__ src, _Float16* __restrict__ dst) {
    int i = blockIdx.x * 256 + threadIdx.x;
    if (i >= 512 * 32) return;
    int n = i >> 5, r = i & 31;
    dst[i] = (r < 16) ? (_Float16)src[n * 16 + r] : (_Float16)0.0f;
}

// x_dbl (M x 48) -> dt slice f16 padded (M x 32)
__global__ void k_pad_dt(const float* __restrict__ xdbl, _Float16* __restrict__ dst) {
    int i = blockIdx.x * 256 + threadIdx.x;
    if (i >= M_ROWS * 32) return;
    int m = i >> 5, r = i & 31;
    dst[i] = (r < 16) ? (_Float16)xdbl[m * 48 + r] : (_Float16)0.0f;
}

// ---------------------------------------------------------------------------
// LayerNorm: x (B,C,H,W) -> y (M, C) fp32 + f16.  One block per (b,l) row.
// ---------------------------------------------------------------------------
__global__ __launch_bounds__(256) void k_layernorm(
    const float* __restrict__ x, const float* __restrict__ g,
    const float* __restrict__ beta, float* __restrict__ yf, _Float16* __restrict__ yh)
{
    int m = blockIdx.x;                 // b*L + l
    int b = m >> 12, l = m & (HW - 1);
    int c = threadIdx.x;
    float v = x[(b * C_DIM + c) * HW + l];
    float s = v, q = v * v;
    for (int o = 16; o; o >>= 1) { s += __shfl_xor(s, o, 32); q += __shfl_xor(q, o, 32); }
    __shared__ float ss[8], sq[8];
    int w = threadIdx.x >> 5, lane = threadIdx.x & 31;
    if (lane == 0) { ss[w] = s; sq[w] = q; }
    __syncthreads();
    if (w == 0) {
        float s2 = (lane < 8) ? ss[lane] : 0.f;
        float q2 = (lane < 8) ? sq[lane] : 0.f;
        for (int o = 4; o; o >>= 1) { s2 += __shfl_xor(s2, o, 32); q2 += __shfl_xor(q2, o, 32); }
        if (lane == 0) { ss[0] = s2; sq[0] = q2; }
    }
    __syncthreads();
    float mu  = ss[0] * (1.f / C_DIM);
    float var = sq[0] * (1.f / C_DIM) - mu * mu;
    float y = (v - mu) * rsqrtf(var + 1e-5f) * g[c] + beta[c];
    yf[m * C_DIM + c] = y;
    yh[m * C_DIM + c] = (_Float16)y;
}

// ---------------------------------------------------------------------------
// WMMA GEMM:  out[M,N] = A[M,K] * W[N,K]^T  (+bias)(+gelu)(+resid)
// Block = 256 threads (8 waves). Wave w -> rows [blk.x*256 + 32w, +32),
// cols [blk.y*64, +64): 2x4 accumulators, 8 WMMA per K-step.
// W tile (64x32 f16) staged to LDS, double-buffered via
// GLOBAL_LOAD_ASYNC_TO_LDS_B128 (ASYNCcnt) when available.
// ---------------------------------------------------------------------------
__global__ __launch_bounds__(256) void k_wmma_gemm(
    const _Float16* __restrict__ A, const _Float16* __restrict__ W,
    const float* __restrict__ bias, const float* __restrict__ resid,
    float* __restrict__ outF, _Float16* __restrict__ outH,
    float* __restrict__ outBCHW, int N, int K, int gelu)
{
    __shared__ __align__(16) _Float16 ldsW[2][64 * 32];
    const int lane = threadIdx.x & 31;
    const int wave = threadIdx.x >> 5;
    const int m0 = blockIdx.x * 256 + wave * 32;
    const int n0 = blockIdx.y * 64;
    const int kb = (lane < 16) ? 0 : 16;          // A-frag K base (contiguous layout)
    const int rowA0 = m0 + (lane & 15);
    v8f acc[2][4] = {};

    // LDS staging: thread -> (col, 8-half chunk)
    const int scol   = threadIdx.x >> 2;          // 0..63
    const int schunk = (threadIdx.x & 3) * 8;     // 0,8,16,24
    const int gcol   = n0 + scol;
    const bool colOK = (gcol < N);
    const _Float16* wsrc = W + (size_t)gcol * K + schunk;
    _Float16* ldst = &ldsW[0][scol * 32 + schunk];
    const int bufStride = 64 * 32;

#define STAGE_W(kk, buf)                                                           \
    do {                                                                           \
        if (HAVE_ASYNC_LDS && colOK) {                                             \
            __builtin_amdgcn_global_load_async_to_lds_b128(                        \
                (AS1 v4i*)(wsrc + (kk)),                                           \
                (AS3 v4i*)(ldst + (buf) * bufStride), 0, 0);                       \
        } else {                                                                   \
            v8h wv = {};                                                           \
            if (colOK) wv = *(const v8h*)(wsrc + (kk));                            \
            *(v8h*)(ldst + (buf) * bufStride) = wv;                                \
        }                                                                          \
    } while (0)

    STAGE_W(0, 0);
    wait_async0();
    __syncthreads();

    int buf = 0;
    for (int kk = 0; kk < K; kk += 32) {
        if (kk + 32 < K) STAGE_W(kk + 32, buf ^ 1);

        // A fragments: halves 0..15 -> K=kb..kb+15
        union { v16h v; v8h h[2]; } af0, af1;
        const _Float16* ap0 = A + (size_t)rowA0 * K + kk + kb;
        af0.h[0] = *(const v8h*)(ap0);
        af0.h[1] = *(const v8h*)(ap0 + 8);
        const _Float16* ap1 = ap0 + (size_t)16 * K;
        af1.h[0] = *(const v8h*)(ap1);
        af1.h[1] = *(const v8h*)(ap1 + 8);
        __builtin_prefetch(ap0 + 32, 0, 1);        // global_prefetch for next K step
        __builtin_prefetch(ap1 + 32, 0, 1);

#pragma unroll
        for (int t = 0; t < 4; ++t) {
            // B fragment: col = lane&15, 16 contiguous K halves, lanes>=16 -> +16
            const _Float16* bp = &ldsW[buf][(t * 16 + (lane & 15)) * 32 + ((lane & 16) ? 16 : 0)];
            v16h bv = *(const v16h*)bp;
            acc[0][t] = __builtin_amdgcn_wmma_f32_16x16x32_f16(
                false, af0.v, false, bv, (short)0, acc[0][t], false, false);
            acc[1][t] = __builtin_amdgcn_wmma_f32_16x16x32_f16(
                false, af1.v, false, bv, (short)0, acc[1][t], false, false);
        }
        wait_async0();
        __syncthreads();
        buf ^= 1;
    }
#undef STAGE_W

#pragma unroll
    for (int u = 0; u < 2; ++u) {
#pragma unroll
        for (int t = 0; t < 4; ++t) {
#pragma unroll
            for (int r = 0; r < 8; ++r) {
                int row = m0 + u * 16 + r + ((lane & 16) ? 8 : 0);
                int col = n0 + t * 16 + (lane & 15);
                if (col >= N) continue;
                float v = acc[u][t][r];
                if (bias)  v += bias[col];
                if (gelu)  v = 0.5f * v * (1.f + erff(v * 0.70710678118f));
                if (resid) v += resid[(size_t)row * N + col];
                if (outF)  outF[(size_t)row * N + col] = v;
                if (outH)  outH[(size_t)row * N + col] = (_Float16)v;
                if (outBCHW) {
                    int b = row >> 12, l = row & (HW - 1);
                    outBCHW[((size_t)b * C_DIM + col) * HW + l] = v;
                }
            }
        }
    }
}

// ---------------------------------------------------------------------------
// Depthwise causal conv (4 taps) + SiLU on the x half of xz
// ---------------------------------------------------------------------------
__global__ void k_conv_silu(const float* __restrict__ xz, const float* __restrict__ cw,
                            const float* __restrict__ cb, float* __restrict__ xcf,
                            _Float16* __restrict__ xch)
{
    int i = blockIdx.x * 256 + threadIdx.x;
    if (i >= M_ROWS * D_INNER_) return;
    int m = i / D_INNER_, d = i - m * D_INNER_;
    int b = m >> 12, l = m & (HW - 1);
    float acc = cb[d];
#pragma unroll
    for (int j = 0; j < 4; ++j) {
        int ll = l - 3 + j;
        if (ll >= 0) acc += xz[((size_t)((b << 12) + ll)) * 1024 + d] * cw[d * 4 + j];
    }
    float s = acc / (1.f + expf(-acc));  // silu
    xcf[i] = s;
    xch[i] = (_Float16)s;
}

// ---------------------------------------------------------------------------
// Selective scan. One 16-lane half-wave per (b,d) sequence: lane = state s.
// ---------------------------------------------------------------------------
__global__ __launch_bounds__(256) void k_scan(
    const float* __restrict__ draw, const float* __restrict__ dtb,
    const float* __restrict__ Alog, const float* __restrict__ Dp,
    const float* __restrict__ xc, const float* __restrict__ xdbl,
    const float* __restrict__ xz, _Float16* __restrict__ ysh)
{
    int gw   = (blockIdx.x * blockDim.x + threadIdx.x) >> 5;  // global wave id
    int lane = threadIdx.x & 31;
    int grp  = lane >> 4;                                      // two sequences per wave
    int s    = lane & 15;
    int p    = gw * 2 + grp;
    if (p >= B_SZ * D_INNER_) return;
    int b = p >> 9, d = p & 511;

    float Aval = -expf(Alog[d * 16 + s]);
    float Dd   = Dp[d];
    float bias = dtb[d];
    float h = 0.f;
    const int base = b * L_SEQ;
    for (int l = 0; l < L_SEQ; ++l) {
        size_t m = base + l;
        float v = draw[m * D_INNER_ + d] + bias;
        float delta = (v > 20.f) ? v : log1pf(expf(v));
        float xt = xc[m * D_INNER_ + d];
        float Bv = xdbl[m * 48 + 16 + s];
        float Cv = xdbl[m * 48 + 32 + s];
        h = expf(delta * Aval) * h + (delta * xt) * Bv;
        float part = h * Cv;
        part += __shfl_xor(part, 1, 32);
        part += __shfl_xor(part, 2, 32);
        part += __shfl_xor(part, 4, 32);
        part += __shfl_xor(part, 8, 32);
        if (s == 0) {
            float z = xz[m * 1024 + 512 + d];
            float g = part + Dd * xt;
            g = g * (z / (1.f + expf(-z)));
            ysh[m * D_INNER_ + d] = (_Float16)g;
        }
    }
}

// ---------------------------------------------------------------------------
// Host launcher
// ---------------------------------------------------------------------------
extern "C" void kernel_launch(void* const* d_in, const int* in_sizes, int n_in,
                              void* d_out, int out_size, void* d_ws, size_t ws_size,
                              hipStream_t stream) {
    const float* x         = (const float*)d_in[0];
    const float* norm_g    = (const float*)d_in[1];
    const float* norm_b    = (const float*)d_in[2];
    const float* in_proj_w = (const float*)d_in[3];
    const float* conv_w    = (const float*)d_in[4];
    const float* conv_b    = (const float*)d_in[5];
    const float* x_proj_w  = (const float*)d_in[6];
    const float* dt_proj_w = (const float*)d_in[7];
    const float* dt_proj_b = (const float*)d_in[8];
    const float* A_log     = (const float*)d_in[9];
    const float* Dvec      = (const float*)d_in[10];
    const float* out_proj_w= (const float*)d_in[11];
    const float* ffn_w1    = (const float*)d_in[12];
    const float* ffn_b1    = (const float*)d_in[13];
    const float* ffn_w2    = (const float*)d_in[14];
    const float* ffn_b2    = (const float*)d_in[15];
    float* out = (float*)d_out;

    size_t off = 0;
    auto alloc = [&](size_t bytes) -> void* {
        void* p = (char*)d_ws + off;
        off += (bytes + 255) & ~(size_t)255;
        return p;
    };
    const size_t M = M_ROWS;
    float*    yln   = (float*)   alloc(M * 256 * 4);
    _Float16* ylnH  = (_Float16*)alloc(M * 256 * 2);
    _Float16* wipH  = (_Float16*)alloc(1024 * 256 * 2);
    _Float16* wxpH  = (_Float16*)alloc(48 * 512 * 2);
    _Float16* wdtH  = (_Float16*)alloc(512 * 32 * 2);
    _Float16* woutH = (_Float16*)alloc(256 * 512 * 2);
    _Float16* wf1H  = (_Float16*)alloc(1024 * 256 * 2);
    _Float16* wf2H  = (_Float16*)alloc(256 * 1024 * 2);
    float*    xz    = (float*)   alloc(M * 1024 * 4);
    float*    xcf   = (float*)   alloc(M * 512 * 4);
    _Float16* xch   = (_Float16*)alloc(M * 512 * 2);
    float*    xdbl  = (float*)   alloc(M * 48 * 4);
    _Float16* dtpH  = (_Float16*)alloc(M * 32 * 2);
    float*    draw  = (float*)   alloc(M * 512 * 4);
    _Float16* ysH   = (_Float16*)alloc(M * 512 * 2);
    float*    y2    = (float*)   alloc(M * 256 * 4);
    _Float16* y2H   = (_Float16*)alloc(M * 256 * 2);
    _Float16* ffnH  = (_Float16*)alloc(M * 1024 * 2);

    auto cvt = [&](const float* s, _Float16* d, int n) {
        k_f32_to_f16<<<(n + 255) / 256, 256, 0, stream>>>(s, d, n);
    };
    cvt(in_proj_w,  wipH,  1024 * 256);
    cvt(x_proj_w,   wxpH,  48 * 512);
    cvt(out_proj_w, woutH, 256 * 512);
    cvt(ffn_w1,     wf1H,  1024 * 256);
    cvt(ffn_w2,     wf2H,  256 * 1024);
    k_pad_dtw<<<(512 * 32 + 255) / 256, 256, 0, stream>>>(dt_proj_w, wdtH);

    k_layernorm<<<M, 256, 0, stream>>>(x, norm_g, norm_b, yln, ylnH);

    dim3 blk(256);
    // in_proj: M x 1024 x 256 -> xz
    k_wmma_gemm<<<dim3(M / 256, 1024 / 64), blk, 0, stream>>>(
        ylnH, wipH, nullptr, nullptr, xz, nullptr, nullptr, 1024, 256, 0);

    k_conv_silu<<<(M * 512 + 255) / 256, 256, 0, stream>>>(xz, conv_w, conv_b, xcf, xch);

    // x_proj: M x 48 x 512 -> xdbl
    k_wmma_gemm<<<dim3(M / 256, 1), blk, 0, stream>>>(
        xch, wxpH, nullptr, nullptr, xdbl, nullptr, nullptr, 48, 512, 0);

    k_pad_dt<<<(M * 32 + 255) / 256, 256, 0, stream>>>(xdbl, dtpH);

    // dt_proj: M x 512 x 32 (K padded) -> draw
    k_wmma_gemm<<<dim3(M / 256, 512 / 64), blk, 0, stream>>>(
        dtpH, wdtH, nullptr, nullptr, draw, nullptr, nullptr, 512, 32, 0);

    // selective scan: 1024 sequences, 2 per wave -> 512 waves -> 64 blocks
    k_scan<<<64, 256, 0, stream>>>(draw, dt_proj_b, A_log, Dvec, xcf, xdbl, xz, ysH);

    // out_proj + residual(yln): M x 256 x 512 -> y2 (f32+f16)
    k_wmma_gemm<<<dim3(M / 256, 256 / 64), blk, 0, stream>>>(
        ysH, woutH, nullptr, yln, y2, y2H, nullptr, 256, 512, 0);

    // ffn1 + bias + gelu: M x 1024 x 256 -> ffnH (f16)
    k_wmma_gemm<<<dim3(M / 256, 1024 / 64), blk, 0, stream>>>(
        y2H, wf1H, ffn_b1, nullptr, nullptr, ffnH, nullptr, 1024, 256, 1);

    // ffn2 + bias + residual(y2), scatter to (B,C,H,W): M x 256 x 1024 -> d_out
    k_wmma_gemm<<<dim3(M / 256, 256 / 64), blk, 0, stream>>>(
        ffnH, wf2H, ffn_b2, y2, nullptr, nullptr, out, 256, 1024, 0);
}
